// MHA_12335146074205
// MI455X (gfx1250) — compile-verified, matrix operand order checked by reading, not run on hardware
//
#include <hip/hip_runtime.h>

// Problem constants (match reference)
constexpr int B_SZ   = 4;
constexpr int SEQ    = 2048;
constexpr int DMODEL = 1024;
constexpr int NHEAD  = 16;
constexpr int HDIM   = 64;

typedef __attribute__((ext_vector_type(16))) __bf16 v16bf;
typedef __attribute__((ext_vector_type(8)))  float  v8f;
typedef int v4i __attribute__((vector_size(16)));

#define AS1 __attribute__((address_space(1)))
#define AS3 __attribute__((address_space(3)))

#if defined(__AMDGCN__) && __has_builtin(__builtin_amdgcn_global_load_async_to_lds_b128)
#define HAVE_ASYNC_LDS 1
#endif

// 16-byte global -> LDS copy. Uses the CDNA5 async-to-LDS path (ASYNCcnt)
// when the builtin exists; synchronous fallback otherwise. Both src and dst
// must be 16-byte aligned.
__device__ __forceinline__ void async_copy16(const void* g, void* l) {
#ifdef HAVE_ASYNC_LDS
  __builtin_amdgcn_global_load_async_to_lds_b128((AS1 v4i*)g, (AS3 v4i*)l, 0, 0);
#else
  *reinterpret_cast<uint4*>(l) = *reinterpret_cast<const uint4*>(g);
#endif
}

__device__ __forceinline__ void async_wait0() {
#ifdef HAVE_ASYNC_LDS
#if __has_builtin(__builtin_amdgcn_s_wait_asynccnt)
  __builtin_amdgcn_s_wait_asynccnt(0);
#else
  asm volatile("s_wait_asynccnt 0x0" ::: "memory");
#endif
#endif
}

// xor-butterfly broadcast within the 32-lane wave; masks <=8 stay inside the
// 16-lane C-matrix row groups. DS_SWIZZLE_B32 (imm pattern, no addr VGPR)
// when available, __shfl_xor fallback.
template <int MASK>
__device__ __forceinline__ float xor_bcast(float x) {
#if defined(__AMDGCN__) && __has_builtin(__builtin_amdgcn_ds_swizzle)
  return __int_as_float(
      __builtin_amdgcn_ds_swizzle(__float_as_int(x), (MASK << 10) | 0x1f));
#else
  return __shfl_xor(x, MASK, 32);
#endif
}

// 16-bit A/B fragment K-pair base for VGPR slot v2 (0..7), lane-half hlf (0/1).
__device__ __forceinline__ int frag_k0(int v2, int hlf) {
  return ((v2 < 4) ? (2 * v2) : (16 + 2 * (v2 - 4))) + hlf * 8;
}

// Load a 16x32 bf16 fragment whose lane-row data is contiguous at rowbase.
__device__ __forceinline__ v16bf load_frag(const __bf16* rowbase, int hlf) {
  v16bf a;
  unsigned int* ap = reinterpret_cast<unsigned int*>(&a);
#pragma unroll
  for (int v2 = 0; v2 < 8; ++v2)
    ap[v2] = *reinterpret_cast<const unsigned int*>(rowbase + frag_k0(v2, hlf));
  return a;
}

// ---------------------------------------------------------------------------
// GEMM: O = X @ W + bias.  X is [M x 1024] (f32 or bf16), W f32 [1024 x 1024].
// OUT_HEAD=true : write bf16 in (B,H,S,64) head-major layout.
// OUT_HEAD=false: write f32 flat [M x 1024].
// Block: 256 threads = 8 waves; tile BM=128, BN=128, BK=32; wave = 64x32.
// ---------------------------------------------------------------------------
template <bool IN_BF16, bool OUT_HEAD>
__global__ __launch_bounds__(256) void proj_gemm(const void* __restrict__ Xv,
                                                 const float* __restrict__ W,
                                                 const float* __restrict__ bias,
                                                 void* __restrict__ Ov, int M) {
  (void)M;
  constexpr int K = 1024, N = 1024;
  constexpr int BM = 128, BN = 128, BK = 32;
  constexpr int AST = BK + 8;  // 40: rows 16B-aligned (80B), bank-stride 20
  constexpr int BST = BK + 2;  // Bst stored transposed: [n][k]
  __shared__ __bf16 As[BM * AST];
  __shared__ __bf16 Bst[BN * BST];

  const int tid = threadIdx.x;
  const int lane = tid & 31;
  const int wave = tid >> 5;   // 0..7
  const int wm = wave >> 2;    // 0..1  (64 rows each)
  const int wn = wave & 3;     // 0..3  (32 cols each)
  const int hlf = lane >> 4;
  const int ln = lane & 15;

  const long row0 = (long)blockIdx.x * BM;
  const int col0 = blockIdx.y * BN;

  v8f acc[4][2];
#pragma unroll
  for (int i = 0; i < 4; ++i)
#pragma unroll
    for (int j = 0; j < 2; ++j) {
      float* p = reinterpret_cast<float*>(&acc[i][j]);
#pragma unroll
      for (int e = 0; e < 8; ++e) p[e] = 0.0f;
    }

  for (int k0t = 0; k0t < K; k0t += BK) {
    __syncthreads();
    // ---- stage X tile (128x32) as bf16 ----
    if (IN_BF16) {
      const __bf16* X = reinterpret_cast<const __bf16*>(Xv);
#pragma unroll
      for (int p = 0; p < 2; ++p) {
        int r = (tid >> 2) + p * 64;
        int c = (tid & 3) * 8;
        async_copy16(X + (row0 + r) * (long)K + k0t + c, &As[r * AST + c]);
      }
    } else {
      const float* X = reinterpret_cast<const float*>(Xv);
#pragma unroll
      for (int p = 0; p < 4; ++p) {
        int r = (tid >> 3) + p * 32;
        int c = (tid & 7) * 4;
        float4 d4 = *reinterpret_cast<const float4*>(X + (row0 + r) * (long)K + k0t + c);
        __bf16* dst = &As[r * AST + c];
        dst[0] = (__bf16)d4.x; dst[1] = (__bf16)d4.y;
        dst[2] = (__bf16)d4.z; dst[3] = (__bf16)d4.w;
      }
    }
    // ---- stage W tile (32x128) transposed -> Bst[n][k], bf16 ----
#pragma unroll
    for (int p = 0; p < 4; ++p) {
      int r = (tid >> 5) + p * 8;   // k within tile
      int c = (tid & 31) * 4;       // n within tile
      float4 d4 = *reinterpret_cast<const float4*>(W + (long)(k0t + r) * N + col0 + c);
      Bst[(c + 0) * BST + r] = (__bf16)d4.x;
      Bst[(c + 1) * BST + r] = (__bf16)d4.y;
      Bst[(c + 2) * BST + r] = (__bf16)d4.z;
      Bst[(c + 3) * BST + r] = (__bf16)d4.w;
    }
    if (IN_BF16) async_wait0();
    __syncthreads();

    // ---- compute: 4 m-subtiles x 2 n-subtiles of WMMA 16x16x32 ----
    v16bf bfr[2];
#pragma unroll
    for (int nn = 0; nn < 2; ++nn)
      bfr[nn] = load_frag(&Bst[(wn * 32 + nn * 16 + ln) * BST], hlf);
#pragma unroll
    for (int mm = 0; mm < 4; ++mm) {
      v16bf afr = load_frag(&As[(wm * 64 + mm * 16 + ln) * AST], hlf);
#pragma unroll
      for (int nn = 0; nn < 2; ++nn)
        acc[mm][nn] = __builtin_amdgcn_wmma_f32_16x16x32_bf16(
            false, afr, false, bfr[nn], (short)0, acc[mm][nn], false, false);
    }
  }

  // ---- epilogue: bias add + store ----
  __bf16* Ob = reinterpret_cast<__bf16*>(Ov);
  float* Of = reinterpret_cast<float*>(Ov);
#pragma unroll
  for (int mm = 0; mm < 4; ++mm) {
    long grow = row0 + wm * 64 + mm * 16;
#pragma unroll
    for (int nn = 0; nn < 2; ++nn) {
      int gcol = col0 + wn * 32 + nn * 16 + ln;
      float bv = bias[gcol];
      const float* av = reinterpret_cast<const float*>(&acc[mm][nn]);
#pragma unroll
      for (int j = 0; j < 8; ++j) {
        long r = grow + j + 8 * hlf;   // C layout: VGPR j -> M=j (+8 upper half)
        float val = av[j] + bv;
        if (OUT_HEAD) {
          long bb = r / SEQ;
          long ss = r % SEQ;
          int h = gcol >> 6, dd = gcol & 63;
          Ob[((bb * NHEAD + h) * SEQ + ss) * HDIM + dd] = (__bf16)val;
        } else {
          Of[r * (long)N + gcol] = val;
        }
      }
    }
  }
}

// ---------------------------------------------------------------------------
// Flash attention over head-major bf16 q/k/v (B,H,S,64).
// Block = 256 threads (8 waves) handles 128 q rows of one (b,h).
// Each wave: 16 q rows, 4 ctx accum frags (16x64), online softmax.
// ctx written bf16 flat (B,S,DMODEL) for the output projection.
// ---------------------------------------------------------------------------
__global__ __launch_bounds__(256) void mha_attn(const __bf16* __restrict__ qh,
                                                const __bf16* __restrict__ kh,
                                                const __bf16* __restrict__ vh,
                                                const int* __restrict__ mask,
                                                __bf16* __restrict__ ctx) {
  constexpr int KST = HDIM + 8;  // 72: rows 16B-aligned (144B), bank-stride 36
  constexpr int VST = 32 + 2;    // Vst[d][key] (transposed V tile)
  constexpr int PST = 32 + 2;    // per-wave P staging [16][32]
  __shared__ __bf16 Ks[32 * KST];
  __shared__ __bf16 Vst[HDIM * VST];
  __shared__ __bf16 Ps[8 * 16 * PST];
  __shared__ int mK[32];
  __shared__ int mQ[128];

  const int tid = threadIdx.x, lane = tid & 31, wave = tid >> 5;
  const int hlf = lane >> 4, ln = lane & 15;
  const int bh = blockIdx.y;           // b*NHEAD + h
  const int bb = bh >> 4, h = bh & 15;
  const int q0 = blockIdx.x * 128;

  const __bf16* qbase = qh + ((long)bh * SEQ + q0) * HDIM;
  const __bf16* kbase = kh + (long)bh * SEQ * HDIM;
  const __bf16* vbase = vh + (long)bh * SEQ * HDIM;

  if (tid < 128) mQ[tid] = mask[bb * SEQ + q0 + tid];
  __syncthreads();

  // Q fragments for this wave's 16 rows (K-dim 64 -> two 16x32 frags).
  const int qrow = wave * 16 + ln;
  v16bf qa[2];
  {
    const __bf16* qp = qbase + (long)qrow * HDIM;
    unsigned int* a0 = reinterpret_cast<unsigned int*>(&qa[0]);
    unsigned int* a1 = reinterpret_cast<unsigned int*>(&qa[1]);
#pragma unroll
    for (int v2 = 0; v2 < 8; ++v2) {
      int k0 = frag_k0(v2, hlf);
      a0[v2] = *reinterpret_cast<const unsigned int*>(qp + k0);
      a1[v2] = *reinterpret_cast<const unsigned int*>(qp + 32 + k0);
    }
  }
  // Precomputed per-row additive mask bias (reference: +(-1e9) where mq&mk).
  float fbq[8];
#pragma unroll
  for (int j = 0; j < 8; ++j)
    fbq[j] = mQ[wave * 16 + j + 8 * hlf] ? -1.0e9f : 0.0f;

  v8f acc[4];
  float mrun[8], lrun[8];
#pragma unroll
  for (int f = 0; f < 4; ++f) {
    float* p = reinterpret_cast<float*>(&acc[f]);
#pragma unroll
    for (int e = 0; e < 8; ++e) p[e] = 0.0f;
  }
#pragma unroll
  for (int j = 0; j < 8; ++j) { mrun[j] = -3.0e38f; lrun[j] = 0.0f; }

  const float scale = 0.125f;  // 1/sqrt(64)

  for (int kt = 0; kt < SEQ; kt += 32) {
    __syncthreads();
    // ---- stage K tile (async global->LDS) and V tile (transposed) ----
    {
      int r = tid >> 3;          // key 0..31
      int c = (tid & 7) * 8;     // d   0..56
      async_copy16(kbase + (long)(kt + r) * HDIM + c, &Ks[r * KST + c]);
      uint4 vd = *reinterpret_cast<const uint4*>(vbase + (long)(kt + r) * HDIM + c);
      const __bf16* ve = reinterpret_cast<const __bf16*>(&vd);
#pragma unroll
      for (int e = 0; e < 8; ++e) Vst[(c + e) * VST + r] = ve[e];
      if (kt + 32 < SEQ) {
        __builtin_prefetch(kbase + (long)(kt + 32 + r) * HDIM + c, 0, 0);
        __builtin_prefetch(vbase + (long)(kt + 32 + r) * HDIM + c, 0, 0);
      }
    }
    if (tid < 32) mK[tid] = mask[bb * SEQ + kt + tid];
    async_wait0();
    __syncthreads();

    // ---- logits = Q @ K^T : two 16x16 C frags (two 16-key subtiles) ----
    v8f lg[2];
#pragma unroll
    for (int sub = 0; sub < 2; ++sub) {
      float* p = reinterpret_cast<float*>(&lg[sub]);
#pragma unroll
      for (int e = 0; e < 8; ++e) p[e] = 0.0f;
      const __bf16* kp = &Ks[(sub * 16 + ln) * KST];  // B row = key, pairs along d
      v16bf b0, b1;
      unsigned int* bp0 = reinterpret_cast<unsigned int*>(&b0);
      unsigned int* bp1 = reinterpret_cast<unsigned int*>(&b1);
#pragma unroll
      for (int v2 = 0; v2 < 8; ++v2) {
        int k0 = frag_k0(v2, hlf);
        bp0[v2] = *reinterpret_cast<const unsigned int*>(kp + k0);
        bp1[v2] = *reinterpret_cast<const unsigned int*>(kp + 32 + k0);
      }
      lg[sub] = __builtin_amdgcn_wmma_f32_16x16x32_bf16(false, qa[0], false, b0,
                                                        (short)0, lg[sub], false, false);
      lg[sub] = __builtin_amdgcn_wmma_f32_16x16x32_bf16(false, qa[1], false, b1,
                                                        (short)0, lg[sub], false, false);
    }

    // ---- scale + mask bias as pure FMA math ----
    float fk0 = (float)mK[ln];
    float fk1 = (float)mK[16 + ln];
    float* l0 = reinterpret_cast<float*>(&lg[0]);
    float* l1 = reinterpret_cast<float*>(&lg[1]);
#pragma unroll
    for (int j = 0; j < 8; ++j) {
      l0[j] = fmaf(l0[j], scale, fbq[j] * fk0);
      l1[j] = fmaf(l1[j], scale, fbq[j] * fk1);
    }

    // ---- online softmax (rows live in 16-lane groups per C VGPR) ----
    float alpha[8];
#pragma unroll
    for (int j = 0; j < 8; ++j) {
      float t = fmaxf(l0[j], l1[j]);
      t = fmaxf(t, xor_bcast<8>(t));
      t = fmaxf(t, xor_bcast<4>(t));
      t = fmaxf(t, xor_bcast<2>(t));
      t = fmaxf(t, xor_bcast<1>(t));
      float mnew = fmaxf(mrun[j], t);
      alpha[j] = __expf(mrun[j] - mnew);
      mrun[j] = mnew;
      float p0 = __expf(l0[j] - mnew);
      float p1 = __expf(l1[j] - mnew);
      l0[j] = p0; l1[j] = p1;
      float rs = p0 + p1;
      rs += xor_bcast<8>(rs);
      rs += xor_bcast<4>(rs);
      rs += xor_bcast<2>(rs);
      rs += xor_bcast<1>(rs);
      lrun[j] = lrun[j] * alpha[j] + rs;
    }
#pragma unroll
    for (int f = 0; f < 4; ++f) {
      float* p = reinterpret_cast<float*>(&acc[f]);
#pragma unroll
      for (int j = 0; j < 8; ++j) p[j] *= alpha[j];
    }

    // ---- re-layout P (C layout) -> A fragment via per-wave LDS ----
    __bf16* pw = &Ps[wave * 16 * PST];
#pragma unroll
    for (int j = 0; j < 8; ++j) {
      int m = j + 8 * hlf;
      pw[m * PST + ln] = (__bf16)l0[j];
      pw[m * PST + 16 + ln] = (__bf16)l1[j];
    }
    __syncthreads();

    v16bf pa = load_frag(&Ps[(wave * 16 + ln) * PST], hlf);
    // ---- ctx += P(16x32) @ V(32x64): 4 d-subtiles ----
#pragma unroll
    for (int f = 0; f < 4; ++f) {
      v16bf vb = load_frag(&Vst[(f * 16 + ln) * VST], hlf);  // row=d, pairs along key
      acc[f] = __builtin_amdgcn_wmma_f32_16x16x32_bf16(false, pa, false, vb,
                                                       (short)0, acc[f], false, false);
    }
  }

  // ---- normalize and store ctx (B,S,DMODEL) bf16 ----
  float rl[8];
#pragma unroll
  for (int j = 0; j < 8; ++j) rl[j] = 1.0f / lrun[j];
#pragma unroll
  for (int f = 0; f < 4; ++f) {
    const float* av = reinterpret_cast<const float*>(&acc[f]);
#pragma unroll
    for (int j = 0; j < 8; ++j) {
      int ss = q0 + wave * 16 + j + 8 * hlf;
      long idx = ((long)bb * SEQ + ss) * DMODEL + h * HDIM + f * 16 + ln;
      ctx[idx] = (__bf16)(av[j] * rl[j]);
    }
  }
}

// ---------------------------------------------------------------------------
extern "C" void kernel_launch(void* const* d_in, const int* in_sizes, int n_in,
                              void* d_out, int out_size, void* d_ws, size_t ws_size,
                              hipStream_t stream) {
  (void)in_sizes; (void)n_in; (void)out_size; (void)ws_size;
  const float* v   = (const float*)d_in[0];
  const float* k   = (const float*)d_in[1];
  const float* q   = (const float*)d_in[2];
  const int*   msk = (const int*)d_in[3];
  const float* Wq  = (const float*)d_in[4];
  const float* bq  = (const float*)d_in[5];
  const float* Wk  = (const float*)d_in[6];
  const float* bk  = (const float*)d_in[7];
  const float* Wv  = (const float*)d_in[8];
  const float* bv  = (const float*)d_in[9];
  const float* Wo  = (const float*)d_in[10];
  const float* bo  = (const float*)d_in[11];

  const size_t elems = (size_t)B_SZ * SEQ * DMODEL;  // 8.39M bf16 each
  __bf16* qhd = (__bf16*)d_ws;
  __bf16* khd = qhd + elems;
  __bf16* vhd = khd + elems;
  __bf16* ctx = vhd + elems;

  const int M = B_SZ * SEQ;          // 8192
  dim3 blk(256);
  dim3 gridP(M / 128, DMODEL / 128); // 64 x 8
  proj_gemm<false, true><<<gridP, blk, 0, stream>>>(q, Wq, bq, qhd, M);
  proj_gemm<false, true><<<gridP, blk, 0, stream>>>(k, Wk, bk, khd, M);
  proj_gemm<false, true><<<gridP, blk, 0, stream>>>(v, Wv, bv, vhd, M);

  dim3 gridA(SEQ / 128, B_SZ * NHEAD);  // 16 x 64
  mha_attn<<<gridA, blk, 0, stream>>>(qhd, khd, vhd, msk, ctx);

  proj_gemm<true, false><<<gridP, blk, 0, stream>>>(ctx, Wo, bo, d_out, M);
}
